// VectorQuantizer_23768349016301
// MI455X (gfx1250) — matrix operand order repaired and checked by reading, not compiled, not run on hardware
//
#include <hip/hip_runtime.h>
#include <hip/hip_bf16.h>

typedef __attribute__((ext_vector_type(16))) __bf16 v16bf;
typedef __attribute__((ext_vector_type(8)))  float  v8f;

// Problem constants
#define VQ_N      131072      // 32*64*64 query vectors
#define VQ_D      64          // embedding dim (== C)
#define VQ_K      1024        // codebook size
#define PLANE     4096        // H*W  (stride between channel planes)
#define IMG       262144      // C*H*W (stride between images)
#define QELEMS    2097152     // 32*64*64*64 output elements
#define NBLOCKS   2048        // VQ_N / 64

// d_ws layout (floats): [0..2047] per-block loss partials, [2048..3071] ||e_k||^2,
// then bf16 E_hi[65536] (128KB) and bf16 E_lo[65536] (128KB)
#define WS_PARTS  0
#define WS_ENORM  2048
#define WS_SPLIT  3072        // float offset where bf16 arrays begin

// ---------------------------------------------------------------------------
// Kernel 1a: codebook row norms ||e_k||^2
// ---------------------------------------------------------------------------
__global__ __launch_bounds__(256) void vq_prep_kernel(const float* __restrict__ emb,
                                                      float* __restrict__ ws) {
    int k = blockIdx.x * 256 + threadIdx.x;
    if (k < VQ_K) {
        const float4* row = (const float4*)(emb + (size_t)k * VQ_D);
        float s = 0.f;
#pragma unroll
        for (int i = 0; i < 16; ++i) {
            float4 t = row[i];
            s += t.x * t.x + t.y * t.y + t.z * t.z + t.w * t.w;
        }
        ws[WS_ENORM + k] = s;
    }
}

// ---------------------------------------------------------------------------
// Kernel 1b: split codebook into bf16 hi/lo (row-major; main kernel's B operand
// takes a contiguous 16-element k run per lane like A, so no repacking needed)
// ---------------------------------------------------------------------------
__global__ __launch_bounds__(256) void vq_split_kernel(const float* __restrict__ emb,
                                                       __bf16* __restrict__ ehi,
                                                       __bf16* __restrict__ elo) {
    int i = blockIdx.x * 256 + threadIdx.x;      // 0 .. 65535
    float v = emb[i];
    __bf16 hi = (__bf16)v;
    ehi[i] = hi;
    elo[i] = (__bf16)(v - (float)hi);
}

// ---------------------------------------------------------------------------
// Kernel 2: main GEMM + argmin + gather + loss partials
//   grid = 2048 blocks x 128 threads (4 waves); wave w owns rows
//   n0 = blockIdx*64 + w*16 of the flattened [N, D] query matrix.
// ---------------------------------------------------------------------------
__global__ __launch_bounds__(128) void vq_main_kernel(const float* __restrict__ x,
                                                      const float* __restrict__ emb,
                                                      const __bf16* __restrict__ ehi,
                                                      const __bf16* __restrict__ elo,
                                                      const float* __restrict__ ws_ro,
                                                      float* __restrict__ out_q,
                                                      float* __restrict__ out_idx,
                                                      float* __restrict__ ws_parts) {
    __shared__ int   sidx[4 * 16];
    __shared__ float swsum[4];

    const int lane = threadIdx.x & 31;
    const int wave = threadIdx.x >> 5;
    const int m    = lane & 15;   // row within A tile / column within C tile
    const int g    = lane >> 4;   // half-wave group

    const int n0 = blockIdx.x * 64 + wave * 16;
    const int b  = n0 >> 12;                       // image index (4096 rows/image)
    const int hw = n0 & 4095;                      // spatial offset within image
    const float* xbase = x + (size_t)b * IMG + hw + m;   // add d*PLANE per component

    const float* enorm = ws_ro + WS_ENORM;

    // ---- Load A tile (16 rows x 64) into WMMA bf16 16x16x32 layout, hi/lo split
    v16bf Ahi[2], Alo[2];
    float xn = 0.f;
#pragma unroll
    for (int ch = 0; ch < 2; ++ch) {
#pragma unroll
        for (int e = 0; e < 16; ++e) {
            int kk = e + 16 * g;                    // K index within 32-chunk
            int d  = 32 * ch + kk;
            float v = xbase[d * PLANE];
            xn += v * v;
            __bf16 hi = (__bf16)v;
            __bf16 lo = (__bf16)(v - (float)hi);
            Ahi[ch][e] = hi;
            Alo[ch][e] = lo;
        }
    }
    // lanes l and l^16 hold disjoint halves of row m's components
    xn += __shfl_xor(xn, 16, 32);                   // xn = ||x_row(m)||^2

    // ---- K loop: 64 tiles of 16 codes, 6 bf16 WMMAs each (hi*hi + hi*lo + lo*hi)
    float bestv[8];
    int   besti[8];
#pragma unroll
    for (int v = 0; v < 8; ++v) { bestv[v] = 3.4e38f; besti[v] = 0; }

    for (int k0 = 0; k0 < VQ_K; k0 += 16) {
        // each lane's B operand = 16 contiguous bf16 (32B) of prepacked row k0+m
        const v16bf* bh = (const v16bf*)(ehi + (size_t)(k0 + m) * VQ_D + 16 * g);
        const v16bf* bl = (const v16bf*)(elo + (size_t)(k0 + m) * VQ_D + 16 * g);
        v8f acc = {};
#pragma unroll
        for (int ch = 0; ch < 2; ++ch) {
            v16bf Bhi = bh[2 * ch];                 // +32 elements per chunk
            v16bf Blo = bl[2 * ch];
            acc = __builtin_amdgcn_wmma_f32_16x16x32_bf16(false, Ahi[ch], false, Bhi,
                                                          (short)0, acc, false, false);
            acc = __builtin_amdgcn_wmma_f32_16x16x32_bf16(false, Ahi[ch], false, Blo,
                                                          (short)0, acc, false, false);
            acc = __builtin_amdgcn_wmma_f32_16x16x32_bf16(false, Alo[ch], false, Bhi,
                                                          (short)0, acc, false, false);
        }
        float en   = enorm[k0 + m];
        int   kidx = k0 + m;
#pragma unroll
        for (int v = 0; v < 8; ++v) {
            float s = fmaf(-2.f, acc[v], en);       // ||e||^2 - 2 x.e
            if (s < bestv[v]) { bestv[v] = s; besti[v] = kidx; }  // strict < keeps first k
        }
    }

    // ---- Butterfly min-reduce across the 16 candidate columns (index tie -> smaller)
#pragma unroll
    for (int off = 8; off >= 1; off >>= 1) {
#pragma unroll
        for (int v = 0; v < 8; ++v) {
            float ov = __shfl_xor(bestv[v], off, 32);
            int   oi = __shfl_xor(besti[v], off, 32);
            if (ov < bestv[v] || (ov == bestv[v] && oi < besti[v])) {
                bestv[v] = ov; besti[v] = oi;
            }
        }
    }
    // all lanes now hold the final (min, argmin) for rows m = v + 8*g

    // ---- Loss partial: ||x - e*||^2 = ||x||^2 + bestv  (per row)
    float lsum = 0.f;
#pragma unroll
    for (int v = 0; v < 8; ++v) {
        float xnr = __shfl(xn, v + 8 * g, 32);      // ||x||^2 of row v+8g
        lsum += xnr + bestv[v];
    }
    if (m != 0) lsum = 0.f;                          // one representative per 16-group
#pragma unroll
    for (int off = 16; off >= 1; off >>= 1) lsum += __shfl_xor(lsum, off, 32);

    if (m == 0) {
#pragma unroll
        for (int v = 0; v < 8; ++v) {
            int row = v + 8 * g;
            sidx[wave * 16 + row] = besti[v];
            out_idx[n0 + row] = (float)besti[v];
        }
    }
    if (lane == 0) swsum[wave] = lsum;
    __syncthreads();

    if (threadIdx.x == 0) {                          // fixed-order, deterministic
        ws_parts[WS_PARTS + blockIdx.x] =
            ((swsum[0] + swsum[1]) + swsum[2]) + swsum[3];
    }

    // ---- Gather codebook rows into [B, C, H, W] output (half-wave coalesced)
    int myidx = sidx[wave * 16 + m];
    const float* eq    = emb  + (size_t)myidx * VQ_D;
    float*       obase = out_q + (size_t)b * IMG + hw + m;
#pragma unroll
    for (int dd = 0; dd < VQ_D; dd += 2) {
        int d = dd + g;
        obase[d * PLANE] = eq[d];
    }
}

// ---------------------------------------------------------------------------
// Kernel 3: deterministic fixed-order loss finalize
// ---------------------------------------------------------------------------
__global__ __launch_bounds__(256) void vq_fin_kernel(const float* __restrict__ ws,
                                                     float* __restrict__ out_loss) {
    __shared__ float red[256];
    int t = threadIdx.x;
    float s = 0.f;
#pragma unroll
    for (int i = 0; i < NBLOCKS / 256; ++i) s += ws[WS_PARTS + t + 256 * i];
    red[t] = s;
    __syncthreads();
    for (int off = 128; off >= 1; off >>= 1) {
        if (t < off) red[t] += red[t + off];
        __syncthreads();
    }
    if (t == 0) out_loss[0] = red[0] * (1.25f / 8388608.f);  // (q + 0.25*e) latent loss
}

extern "C" void kernel_launch(void* const* d_in, const int* in_sizes, int n_in,
                              void* d_out, int out_size, void* d_ws, size_t ws_size,
                              hipStream_t stream) {
    (void)in_sizes; (void)n_in; (void)out_size; (void)ws_size;
    const float* x   = (const float*)d_in[0];   // [32, 64, 64, 64]
    const float* emb = (const float*)d_in[1];   // [1024, 64]
    float* out      = (float*)d_out;
    float* out_q    = out;                      // 2097152 floats
    float* out_loss = out + QELEMS;             // 1 float
    float* out_idx  = out + QELEMS + 1;         // 131072 floats (indices)
    float*  ws  = (float*)d_ws;
    __bf16* ehi = (__bf16*)(ws + WS_SPLIT);             // 65536 bf16 (128KB)
    __bf16* elo = (__bf16*)(ws + WS_SPLIT) + VQ_K * VQ_D;

    vq_prep_kernel<<<4, 256, 0, stream>>>(emb, ws);
    vq_split_kernel<<<VQ_K * VQ_D / 256, 256, 0, stream>>>(emb, ehi, elo);
    vq_main_kernel<<<NBLOCKS, 128, 0, stream>>>(x, emb, ehi, elo, ws, out_q, out_idx, ws);
    vq_fin_kernel<<<1, 256, 0, stream>>>(ws, out_loss);
}
